// EdgeFeatureGNNLayer_34230889349205
// MI455X (gfx1250) — matrix operand order repaired and checked by reading, not compiled, not run on hardware
//
#include <hip/hip_runtime.h>
#include <hip/hip_bf16.h>

// EdgeFeatureGNNLayer for MI455X (gfx1250, wave32, WMMA).
//   h = node_feats @ Wn + bn          [100000,128]
//   e = edge_feats @ We + be          [1600000,128]
//   m = h[src] + e ; out = segment_sum(m, dst)
// Memory-bound (~0.5 GB HBM min at 23.3 TB/s => ~22us floor); gather/scatter
// targets (51.2 MB each) are L2-resident (192 MB L2). bf16 WMMA, f32 acc.
// LDS K-strides padded (136 / 72 ushorts) for conflict-free fragment reads
// while staying 16B-aligned so ds_load_b128 vectorization survives.

#define N_NODES 100000
#define N_EDGES 1600000
#define NODE_IN 128
#define EDGE_IN 64
#define OUT_F   128

#define KS_N 136   // padded K-stride (ushorts) for K=128 tiles: 272B = 17*16
#define KS_E 72    // padded K-stride (ushorts) for K=64  tiles: 144B = 9*16

typedef __attribute__((ext_vector_type(16))) __bf16        v16bf;
typedef __attribute__((ext_vector_type(8)))  float         v8f;
typedef __attribute__((ext_vector_type(8)))  unsigned int  v8u;

union FragU { v8u u; v16bf b; };

__device__ __forceinline__ unsigned short f32_to_bf16(float f) {
    unsigned int x = __float_as_uint(f);
    x += 0x7FFFu + ((x >> 16) & 1u);   // round to nearest even
    return (unsigned short)(x >> 16);
}

__device__ __forceinline__ unsigned int pack_bf16x2(float lo, float hi) {
    return (unsigned int)f32_to_bf16(lo) | ((unsigned int)f32_to_bf16(hi) << 16);
}

// A-fragment (16x32 bf16, ISA 7.12.2): lane holds row M=lane&15; element pairs
// at K = ((vg>>2)<<4)+(hi<<3)+((vg&3)<<1). base: lane's row start, 16B-aligned.
__device__ __forceinline__ v16bf build_frag_a(const unsigned short* base, int hi) {
    FragU t;
#pragma unroll
    for (int vg = 0; vg < 8; ++vg) {
        int k0 = ((vg >> 2) << 4) + (hi << 3) + ((vg & 3) << 1);
        t.u[vg] = *(const unsigned int*)(base + k0);
    }
    return t.b;
}

// B-fragment (32x16 bf16): lane holds column N=lane&15; element pairs at
// K = ((vg>>2)<<4)+(hi<<3)+((vg&3)<<1), matching the A-fragment K order.
__device__ __forceinline__ v16bf build_frag_b(const unsigned short* base, int hi) {
    FragU t;
#pragma unroll
    for (int vg = 0; vg < 8; ++vg) {
        int k0 = ((vg >> 2) << 4) + (hi << 3) + ((vg & 3) << 1);
        t.u[vg] = *(const unsigned int*)(base + k0);
    }
    return t.b;
}

__global__ void zero_f32_kernel(float* __restrict__ p, int count) {
    int i = blockIdx.x * blockDim.x + threadIdx.x;
    if (i < count) p[i] = 0.0f;
}

// ---------------------------------------------------------------------------
// Kernel 1: h = node_feats @ Wn + bn.  1 wave = one 16-row tile, full 128 cols.
// ---------------------------------------------------------------------------
__global__ __launch_bounds__(128) void node_gemm_kernel(
    const float* __restrict__ node_feats, const float* __restrict__ Wn,
    const float* __restrict__ bn, float* __restrict__ h)
{
    __shared__ __align__(16) unsigned short WnT[OUT_F * KS_N];   // [col][k] ~34KB
    __shared__ __align__(16) unsigned short Ast[4][16 * KS_N];   // per-wave ~4.25KB

    const int tid  = threadIdx.x;
    const int wave = tid >> 5;
    const int lane = tid & 31;
    const int hi   = lane >> 4;
    const int n    = lane & 15;

    // Stage Wn transposed as bf16: WnT[col][k] = Wn[k][col].
    // Per-wave banks: 32 consecutive cols * 4-bank shift -> <=2-way conflict.
    for (int i = tid; i < NODE_IN * OUT_F; i += 128) {
        int k   = i >> 7;     // Wn row (0..127)
        int col = i & 127;
        WnT[col * KS_N + k] = f32_to_bf16(Wn[i]);
    }

    const int  tile  = blockIdx.x * 4 + wave;        // 16-row tile id
    const bool valid = (tile * 16) < N_NODES;        // N_NODES % 16 == 0

    if (valid) {
        // Stage A tile 16x128 as bf16: float2 coalesced reads, packed u32 writes.
        const float* arow = node_feats + tile * 16 * NODE_IN;
#pragma unroll
        for (int j = 0; j < 32; ++j) {               // 1024 u32 / 32 lanes
            int p  = lane + j * 32;                  // pair index
            int r  = p >> 6;                         // (p*2)/128
            int c  = (p & 63) * 2;
            float2 v = *(const float2*)(arow + r * NODE_IN + c);
            *(unsigned int*)&Ast[wave][r * KS_N + c] = pack_bf16x2(v.x, v.y);
        }
    }
    __syncthreads();
    if (!valid) return;

    const unsigned short* Arow = &Ast[wave][(lane & 15) * KS_N];

#pragma unroll
    for (int nt = 0; nt < 8; ++nt) {
        float bias = bn[nt * 16 + n];
        v8f acc = {bias, bias, bias, bias, bias, bias, bias, bias};
        const unsigned short* Bcol = &WnT[(nt * 16 + n) * KS_N];
#pragma unroll
        for (int kt = 0; kt < 4; ++kt) {
            v16bf a = build_frag_a(Arow + kt * 32, hi);
            v16bf b = build_frag_b(Bcol + kt * 32, hi);
            acc = __builtin_amdgcn_wmma_f32_16x16x32_bf16(
                      false, a, false, b, (short)0, acc, false, false);
        }
#pragma unroll
        for (int r = 0; r < 8; ++r) {
            int row = tile * 16 + r + 8 * hi;        // C/D layout: M = r + 8*hi
            h[row * OUT_F + nt * 16 + n] = acc[r];
        }
    }
}

// ---------------------------------------------------------------------------
// Kernel 2: e = edge_feats @ We + be, fused with m = e + h[src] and
// atomic scatter-sum into out[dst]. 1 wave = one 16-edge tile.
// ---------------------------------------------------------------------------
__global__ __launch_bounds__(256) void edge_gemm_scatter_kernel(
    const float* __restrict__ edge_feats, const float* __restrict__ We,
    const float* __restrict__ be, const int* __restrict__ src,
    const int* __restrict__ dst, const float* __restrict__ h,
    float* __restrict__ out)
{
    __shared__ __align__(16) unsigned short WeT[OUT_F * KS_E];   // [col][k] ~18KB
    __shared__ __align__(16) unsigned short Ast[8][16 * KS_E];   // per-wave ~2.25KB
    __shared__ __align__(16) int SD_lds[8][32];                  // [wave][0:16)=src, [16:32)=dst

    const int tid  = threadIdx.x;
    const int wave = tid >> 5;
    const int lane = tid & 31;
    const int hi   = lane >> 4;
    const int n    = lane & 15;

    const int tile = blockIdx.x * 8 + wave;          // grid sized exactly: always valid

    // Async-copy the 16 src + 16 dst indices straight into LDS (ASYNCcnt path).
    {
        unsigned lds_addr = (unsigned)(size_t)&SD_lds[wave][lane];
        unsigned voff     = (unsigned)((tile * 16 + n) * sizeof(int));
        if (lane < 16) {
            asm volatile("global_load_async_to_lds_b32 %0, %1, %2"
                         :: "v"(lds_addr), "v"(voff), "s"(src) : "memory");
        } else {
            asm volatile("global_load_async_to_lds_b32 %0, %1, %2"
                         :: "v"(lds_addr), "v"(voff), "s"(dst) : "memory");
        }
    }

    // Stage We transposed as bf16: WeT[col][k] = We[k][col]
    for (int i = tid; i < EDGE_IN * OUT_F; i += 256) {
        int k   = i >> 7;     // We row (0..63)
        int col = i & 127;
        WeT[col * KS_E + k] = f32_to_bf16(We[i]);
    }

    // Stage edge-feature tile 16x64 as bf16: float2 reads, packed u32 writes.
    {
        const float* arow = edge_feats + tile * 16 * EDGE_IN;
#pragma unroll
        for (int j = 0; j < 16; ++j) {               // 512 u32 / 32 lanes
            int p  = lane + j * 32;
            int r  = p >> 5;                         // (p*2)/64
            int c  = (p & 31) * 2;
            float2 v = *(const float2*)(arow + r * EDGE_IN + c);
            *(unsigned int*)&Ast[wave][r * KS_E + c] = pack_bf16x2(v.x, v.y);
        }
    }

    asm volatile("s_wait_asynccnt 0x0" ::: "memory");  // our wave's SD_lds ready
    __syncthreads();                                   // WeT ready block-wide

    const unsigned short* Arow = &Ast[wave][(lane & 15) * KS_E];

#pragma unroll
    for (int nt = 0; nt < 8; ++nt) {
        float bias = be[nt * 16 + n];
        v8f acc = {bias, bias, bias, bias, bias, bias, bias, bias};
        const unsigned short* Bcol = &WeT[(nt * 16 + n) * KS_E];
#pragma unroll
        for (int kt = 0; kt < 2; ++kt) {
            v16bf a = build_frag_a(Arow + kt * 32, hi);
            v16bf b = build_frag_b(Bcol + kt * 32, hi);
            acc = __builtin_amdgcn_wmma_f32_16x16x32_bf16(
                      false, a, false, b, (short)0, acc, false, false);
        }
        const int col = nt * 16 + n;
#pragma unroll
        for (int r = 0; r < 8; ++r) {
            int m = r + 8 * hi;                      // edge within tile
            int s = SD_lds[wave][m];
            int d = SD_lds[wave][16 + m];
            // lanes 0-15 touch 64 contiguous bytes of one row -> L2-friendly
            float val = acc[r] + h[s * OUT_F + col];
            atomicAdd(&out[d * OUT_F + col], val);
        }
    }
}

extern "C" void kernel_launch(void* const* d_in, const int* in_sizes, int n_in,
                              void* d_out, int out_size, void* d_ws, size_t ws_size,
                              hipStream_t stream)
{
    const float* node_feats = (const float*)d_in[0];
    const float* edge_feats = (const float*)d_in[1];
    const int*   src        = (const int*)d_in[2];
    const int*   dst        = (const int*)d_in[3];
    const float* Wn         = (const float*)d_in[4];
    const float* bn         = (const float*)d_in[5];
    const float* We         = (const float*)d_in[6];
    const float* be         = (const float*)d_in[7];
    float* out = (float*)d_out;
    float* h   = (float*)d_ws;                       // 100000*128 f32 = 51.2 MB

    const int total = N_NODES * OUT_F;
    zero_f32_kernel<<<(total + 1023) / 1024, 1024, 0, stream>>>(out, total);

    const int node_tiles = N_NODES / 16;             // 6250
    node_gemm_kernel<<<(node_tiles + 3) / 4, 128, 0, stream>>>(node_feats, Wn, bn, h);

    const int edge_tiles = N_EDGES / 16;             // 100000 = 12500 * 8
    edge_gemm_scatter_kernel<<<edge_tiles / 8, 256, 0, stream>>>(
        edge_feats, We, be, src, dst, h, out);
}
